// RelativeGlobalAttention_9320079032917
// MI455X (gfx1250) — matrix operand order repaired and checked by reading, not compiled, not run on hardware
//
#include <hip/hip_runtime.h>

// ---------- types ----------
typedef __attribute__((ext_vector_type(16))) __bf16 v16bf;
typedef __attribute__((ext_vector_type(8)))  float  v8f;

union FragU { uint4 u[2]; v16bf v; };

__device__ __forceinline__ unsigned short f2bf(float f) {
  unsigned int u = __float_as_uint(f);
  u += 0x7FFFu + ((u >> 16) & 1u);   // round-to-nearest-even
  return (unsigned short)(u >> 16);
}

__device__ __forceinline__ v8f wmma_bf16(v16bf a, v16bf b, v8f c) {
  return __builtin_amdgcn_wmma_f32_16x16x32_bf16(false, a, false, b, (short)0, c, false, false);
}

// ---- CDNA5 async global->LDS copy (ASYNCcnt path, ISA 15.18 op 98) ----
__device__ __forceinline__ void async_g2l_b128(void* lds_dst, const void* gsrc) {
  asm volatile("global_load_async_to_lds_b128 %0, %1, off"
               :
               : "v"((unsigned)(size_t)lds_dst), "v"(gsrc)
               : "memory");
}
__device__ __forceinline__ void wait_async0() {
  asm volatile("s_wait_asynccnt 0x0" ::: "memory");
}
__device__ __forceinline__ void wait_async4() {   // leave newest 4 in flight
  asm volatile("s_wait_asynccnt 0x4" ::: "memory");
}

// A fragment: 16x32 bf16, row-major LDS [row][k] (ldk elements).
__device__ __forceinline__ v16bf frag_ld_a(const unsigned short* lds, int row0, int k0, int ldk, int lane) {
  int m  = row0 + (lane & 15);
  int kh = (lane >> 4) << 3;
  const unsigned short* base = lds + m * ldk + k0 + kh;
  FragU f;
  f.u[0] = *(const uint4*)(base);
  f.u[1] = *(const uint4*)(base + 16);
  return f.v;
}

// B fragment: 32x16 bf16, LDS [col][k] (ldk elements).
__device__ __forceinline__ v16bf frag_ld_b(const unsigned short* lds, int col0, int k0, int ldk, int lane) {
  int n = col0 + (lane & 15);
  const unsigned short* base = lds + n * ldk + k0 + ((lane >> 4) << 3);
  FragU f;
  f.u[0] = *(const uint4*)(base);
  f.u[1] = *(const uint4*)(base + 16);
  return f.v;
}

// ---------- kernel 0a: f32 -> bf16 elementwise (8 per thread) ----------
__global__ __launch_bounds__(256) void k_cvt_bf16(const float* __restrict__ in,
                                                  unsigned short* __restrict__ out, int n8) {
  int i = blockIdx.x * 256 + threadIdx.x;
  if (i >= n8) return;
  const float4* p = (const float4*)in + (size_t)i * 2;
  float4 a = p[0], b = p[1];
  uint4 o;
  o.x = (unsigned)f2bf(a.x) | ((unsigned)f2bf(a.y) << 16);
  o.y = (unsigned)f2bf(a.z) | ((unsigned)f2bf(a.w) << 16);
  o.z = (unsigned)f2bf(b.x) | ((unsigned)f2bf(b.y) << 16);
  o.w = (unsigned)f2bf(b.z) | ((unsigned)f2bf(b.w) << 16);
  ((uint4*)out)[i] = o;
}

// ---------- kernel 0b: transpose+convert weight [K][N] f32 -> [N][K] bf16 ----------
__global__ __launch_bounds__(256) void k_transpose_bf16(const float* __restrict__ in,
                                                        unsigned short* __restrict__ out,
                                                        int Kd, int Nd) {
  int i = blockIdx.x * 256 + threadIdx.x;
  if (i >= Kd * Nd) return;
  int n = i / Kd, k = i - n * Kd;
  out[i] = f2bf(in[(size_t)k * Nd + n]);
}

// ---------- kernel 1: QKV GEMM  [92416,512]bf16 x [1536,512]bf16^T -> Q/K/Vt ----------
__global__ __launch_bounds__(256) void k_qkv_gemm(const unsigned short* __restrict__ Xb,
                                                  const unsigned short* __restrict__ WbT,
                                                  unsigned short* __restrict__ Q,
                                                  unsigned short* __restrict__ Kt,
                                                  unsigned short* __restrict__ Vt) {
  __shared__ unsigned short As[2][128 * 32];
  __shared__ unsigned short Bs[2][128 * 32];
  const int tid = threadIdx.x;
  const int lane = tid & 31, w = tid >> 5;
  const int wm = w & 3, wn = w >> 2;
  const int m0 = blockIdx.y * 128;
  const int n0 = blockIdx.x * 128;

  v8f acc[2][4];
#pragma unroll
  for (int mt = 0; mt < 2; ++mt)
#pragma unroll
    for (int nt = 0; nt < 4; ++nt)
#pragma unroll
      for (int r = 0; r < 8; ++r) acc[mt][nt][r] = 0.f;

  const int lr = tid >> 1;
  const int lc = (tid & 1) * 16;
  const unsigned short* gA = Xb  + (size_t)(m0 + lr) * 512 + lc;
  const unsigned short* gB = WbT + (size_t)(n0 + lr) * 512 + lc;
  const int ldsoff = lr * 32 + lc;

  // prologue: tile 0 -> buffer 0
  async_g2l_b128(As[0] + ldsoff,     gA);
  async_g2l_b128(As[0] + ldsoff + 8, gA + 8);
  async_g2l_b128(Bs[0] + ldsoff,     gB);
  async_g2l_b128(Bs[0] + ldsoff + 8, gB + 8);

  for (int it = 0; it < 16; ++it) {
    const int cur = it & 1, nxt = cur ^ 1;
    __syncthreads();                       // readers of buf nxt (iter it-1) done
    if (it + 1 < 16) {
      const int k1 = (it + 1) * 32;
      async_g2l_b128(As[nxt] + ldsoff,     gA + k1);
      async_g2l_b128(As[nxt] + ldsoff + 8, gA + k1 + 8);
      async_g2l_b128(Bs[nxt] + ldsoff,     gB + k1);
      async_g2l_b128(Bs[nxt] + ldsoff + 8, gB + k1 + 8);
      if (it + 2 < 16) __builtin_prefetch(gA + (it + 2) * 32, 0, 1);
      wait_async4();                       // buf cur landed; buf nxt in flight
    } else {
      wait_async0();
    }
    __syncthreads();                       // all waves' buf cur ready
    v16bf a0 = frag_ld_a(As[cur], wm * 32,      0, 32, lane);
    v16bf a1 = frag_ld_a(As[cur], wm * 32 + 16, 0, 32, lane);
#pragma unroll
    for (int nt = 0; nt < 4; ++nt) {
      v16bf bf = frag_ld_b(Bs[cur], wn * 64 + nt * 16, 0, 32, lane);
      acc[0][nt] = wmma_bf16(a0, bf, acc[0][nt]);
      acc[1][nt] = wmma_bf16(a1, bf, acc[1][nt]);
    }
  }

  const int hi8 = (lane >> 4) << 3;
  const int nl  = lane & 15;
#pragma unroll
  for (int mt = 0; mt < 2; ++mt) {
#pragma unroll
    for (int nt = 0; nt < 4; ++nt) {
#pragma unroll
      for (int r = 0; r < 8; ++r) {
        int m = m0 + wm * 32 + mt * 16 + r + hi8;
        int n = n0 + wn * 64 + nt * 16 + nl;
        int bi = m / 361, tok = m - bi * 361;
        int part = n >> 9, c = n & 511;
        int hh = c >> 6, dh = c & 63;
        float v = acc[mt][nt][r];
        size_t bh = (size_t)(bi * 8 + hh);
        if (part == 0)      Q [(bh * 361 + tok) * 64 + dh] = f2bf(v * 0.125f); // *Dh^-0.5
        else if (part == 1) Kt[(bh * 361 + tok) * 64 + dh] = f2bf(v);
        else                Vt[(bh * 64 + dh) * 384 + tok] = f2bf(v);          // V transposed, padded
      }
    }
  }
}

// ---------- kernel 2: attention with LDS-resident scores ----------
// grid (6 qtiles, 8 heads, 256 batch), 256 threads, ~254KB dynamic LDS
#define ATTN_SMEM 259712
__global__ __launch_bounds__(256) void k_attn(const unsigned short* __restrict__ Qg,
                                              const unsigned short* __restrict__ Kg,
                                              const unsigned short* __restrict__ Vtg,
                                              const float* __restrict__ bias_table,
                                              unsigned short* __restrict__ Og) {
  extern __shared__ char smem[];
  unsigned short* Qs  = (unsigned short*)(smem);           // 64 x 64
  unsigned short* Ks  = (unsigned short*)(smem + 8192);    // 384 x 64
  unsigned short* Vts = (unsigned short*)(smem + 57344);   // 64 x 384
  float*          Sf  = (float*)        (smem + 106496);   // 64 x 384
  unsigned short* Pb  = (unsigned short*)(smem + 204800);  // 64 x 384
  float*          biasL = (float*)      (smem + 253952);   // 1369
  float*          rsum  = (float*)      (smem + 259456);   // 64

  const int tid = threadIdx.x;
  const int qt = blockIdx.x, hh = blockIdx.y, bb = blockIdx.z;
  const int qbase = qt * 64;
  const size_t bh = (size_t)(bb * 8 + hh);

  // cooperative fills via async global->LDS b128 (zero-pad OOB rows with ds stores)
  const uint4 z4 = make_uint4(0, 0, 0, 0);
  for (int i = tid; i < 512; i += 256) {             // Q tile: 64 rows x 8 chunks
    int row = i >> 3, ch = i & 7;
    int tok = qbase + row;
    if (tok < 361) async_g2l_b128(Qs + row * 64 + ch * 8, Qg + (bh * 361 + tok) * 64 + ch * 8);
    else           *((uint4*)(Qs + row * 64) + ch) = z4;
  }
  for (int i = tid; i < 3072; i += 256) {            // K tile: 384 rows x 8 chunks
    int row = i >> 3, ch = i & 7;
    if (row < 361) async_g2l_b128(Ks + row * 64 + ch * 8, Kg + (bh * 361 + row) * 64 + ch * 8);
    else           *((uint4*)(Ks + row * 64) + ch) = z4;
  }
  for (int i = tid; i < 3072; i += 256) {            // V^T tile: 64 rows x 48 chunks (padded stride)
    int dh = i / 48, ch = i - dh * 48;
    async_g2l_b128(Vts + dh * 384 + ch * 8, Vtg + (bh * 64 + dh) * 384 + ch * 8);
  }
  for (int i = tid; i < 1369; i += 256) biasL[i] = bias_table[i * 8 + hh];
  wait_async0();
  __syncthreads();
  for (int i = tid; i < 64 * 23; i += 256) {         // zero V^T pad keys 361..383
    int dh = i / 23, kk = 361 + (i - dh * 23);
    Vts[dh * 384 + kk] = 0;
  }
  __syncthreads();

  const int lane = tid & 31, w = tid >> 5;
  const int qt2 = w & 3, kg = w >> 2;                 // 4 q-subtiles x 2 key-halves
  const int hi8 = (lane >> 4) << 3, nl = lane & 15;

  // ---- S = Q K^T + bias ----
  v16bf a0 = frag_ld_a(Qs, qt2 * 16, 0,  64, lane);
  v16bf a1 = frag_ld_a(Qs, qt2 * 16, 32, 64, lane);
#pragma unroll
  for (int ntt = 0; ntt < 12; ++ntt) {
    int nt = kg * 12 + ntt;
    v8f acc;
#pragma unroll
    for (int r = 0; r < 8; ++r) acc[r] = 0.f;
    acc = wmma_bf16(a0, frag_ld_b(Ks, nt * 16, 0,  64, lane), acc);
    acc = wmma_bf16(a1, frag_ld_b(Ks, nt * 16, 32, 64, lane), acc);
    int j = nt * 16 + nl;
    int jc = j < 361 ? j : 360;
    int jy = jc / 19, jx = jc - jy * 19;
#pragma unroll
    for (int r = 0; r < 8; ++r) {
      int irow = qt2 * 16 + r + hi8;
      int i = qbase + irow;
      int ic = i < 361 ? i : 360;
      int iy = ic / 19, ix = ic - iy * 19;
      int bidx = (iy - jy + 18) * 37 + (ix - jx + 18);
      Sf[irow * 384 + j] = acc[r] + biasL[bidx];
    }
  }
  __syncthreads();

  // ---- softmax rows (unnormalized exp; normalize in O epilogue) ----
  if (tid < 64) {
    int row = tid;
    float mx = -1e30f;
    for (int c = 0; c < 361; ++c) mx = fmaxf(mx, Sf[row * 384 + c]);
    float s = 0.f;
    for (int c = 0; c < 361; ++c) {
      float e = __expf(Sf[row * 384 + c] - mx);
      s += e;
      Pb[row * 384 + c] = f2bf(e);
    }
    for (int c = 361; c < 384; ++c) Pb[row * 384 + c] = 0;
    rsum[row] = s;
  }
  __syncthreads();

  // ---- O = P V ----
  v8f o0, o1;
#pragma unroll
  for (int r = 0; r < 8; ++r) { o0[r] = 0.f; o1[r] = 0.f; }
#pragma unroll
  for (int kt = 0; kt < 12; ++kt) {
    v16bf ap = frag_ld_a(Pb, qt2 * 16, kt * 32, 384, lane);
    o0 = wmma_bf16(ap, frag_ld_b(Vts, (kg * 2 + 0) * 16, kt * 32, 384, lane), o0);
    o1 = wmma_bf16(ap, frag_ld_b(Vts, (kg * 2 + 1) * 16, kt * 32, 384, lane), o1);
  }
#pragma unroll
  for (int r = 0; r < 8; ++r) {
    int irow = qt2 * 16 + r + hi8;
    int tok = qbase + irow;
    if (tok < 361) {
      float inv = 1.f / rsum[irow];
      size_t o = ((size_t)bb * 361 + tok) * 512 + hh * 64;
      Og[o + (kg * 2 + 0) * 16 + nl] = f2bf(o0[r] * inv);
      Og[o + (kg * 2 + 1) * 16 + nl] = f2bf(o1[r] * inv);
    }
  }
}

// ---------- kernel 3: proj GEMM  out = O @ proj_w + b  (f32 out) ----------
__global__ __launch_bounds__(256) void k_proj_gemm(const unsigned short* __restrict__ Ob,
                                                   const unsigned short* __restrict__ PbT,
                                                   const float* __restrict__ pbias,
                                                   float* __restrict__ out) {
  __shared__ unsigned short As[2][128 * 32];
  __shared__ unsigned short Bs[2][128 * 32];
  const int tid = threadIdx.x;
  const int lane = tid & 31, w = tid >> 5;
  const int wm = w & 3, wn = w >> 2;
  const int m0 = blockIdx.y * 128;
  const int n0 = blockIdx.x * 128;

  v8f acc[2][4];
#pragma unroll
  for (int mt = 0; mt < 2; ++mt)
#pragma unroll
    for (int nt = 0; nt < 4; ++nt)
#pragma unroll
      for (int r = 0; r < 8; ++r) acc[mt][nt][r] = 0.f;

  const int lr = tid >> 1;
  const int lc = (tid & 1) * 16;
  const unsigned short* gA = Ob  + (size_t)(m0 + lr) * 512 + lc;
  const unsigned short* gB = PbT + (size_t)(n0 + lr) * 512 + lc;
  const int ldsoff = lr * 32 + lc;

  async_g2l_b128(As[0] + ldsoff,     gA);
  async_g2l_b128(As[0] + ldsoff + 8, gA + 8);
  async_g2l_b128(Bs[0] + ldsoff,     gB);
  async_g2l_b128(Bs[0] + ldsoff + 8, gB + 8);

  for (int it = 0; it < 16; ++it) {
    const int cur = it & 1, nxt = cur ^ 1;
    __syncthreads();
    if (it + 1 < 16) {
      const int k1 = (it + 1) * 32;
      async_g2l_b128(As[nxt] + ldsoff,     gA + k1);
      async_g2l_b128(As[nxt] + ldsoff + 8, gA + k1 + 8);
      async_g2l_b128(Bs[nxt] + ldsoff,     gB + k1);
      async_g2l_b128(Bs[nxt] + ldsoff + 8, gB + k1 + 8);
      if (it + 2 < 16) __builtin_prefetch(gA + (it + 2) * 32, 0, 1);
      wait_async4();
    } else {
      wait_async0();
    }
    __syncthreads();
    v16bf a0 = frag_ld_a(As[cur], wm * 32,      0, 32, lane);
    v16bf a1 = frag_ld_a(As[cur], wm * 32 + 16, 0, 32, lane);
#pragma unroll
    for (int nt = 0; nt < 4; ++nt) {
      v16bf bf = frag_ld_b(Bs[cur], wn * 64 + nt * 16, 0, 32, lane);
      acc[0][nt] = wmma_bf16(a0, bf, acc[0][nt]);
      acc[1][nt] = wmma_bf16(a1, bf, acc[1][nt]);
    }
  }

  const int hi8 = (lane >> 4) << 3;
  const int nl  = lane & 15;
#pragma unroll
  for (int mt = 0; mt < 2; ++mt) {
#pragma unroll
    for (int nt = 0; nt < 4; ++nt) {
#pragma unroll
      for (int r = 0; r < 8; ++r) {
        int m = m0 + wm * 32 + mt * 16 + r + hi8;
        int n = n0 + wn * 64 + nt * 16 + nl;
        out[(size_t)m * 512 + n] = acc[mt][nt][r] + pbias[n];
      }
    }
  }
}

// ---------- launch ----------
extern "C" void kernel_launch(void* const* d_in, const int* in_sizes, int n_in,
                              void* d_out, int out_size, void* d_ws, size_t ws_size,
                              hipStream_t stream) {
  const float* x          = (const float*)d_in[0];
  const float* qkv_w      = (const float*)d_in[1];
  const float* proj_w     = (const float*)d_in[2];
  const float* proj_b     = (const float*)d_in[3];
  const float* bias_table = (const float*)d_in[4];
  float* out = (float*)d_out;
  char* ws = (char*)d_ws;

  size_t off = 0;
  unsigned short* Xb  = (unsigned short*)(ws + off); off += (size_t)92416 * 512 * 2;      // x bf16
  unsigned short* WbT = (unsigned short*)(ws + off); off += (size_t)1536 * 512 * 2;       // qkv_w^T bf16
  unsigned short* PbT = (unsigned short*)(ws + off); off += (size_t)512 * 512 * 2;        // proj_w^T bf16
  unsigned short* Q   = (unsigned short*)(ws + off); off += (size_t)256 * 8 * 361 * 64 * 2;
  unsigned short* K   = (unsigned short*)(ws + off); off += (size_t)256 * 8 * 361 * 64 * 2;
  unsigned short* Vt  = (unsigned short*)(ws + off); off += (size_t)256 * 8 * 64 * 384 * 2; // V^T, padded
  unsigned short* O   = (unsigned short*)(ws + off); off += (size_t)92416 * 512 * 2;

  k_cvt_bf16<<<23104, 256, 0, stream>>>(x, Xb, 5914624);            // 92416*512/8
  k_transpose_bf16<<<3072, 256, 0, stream>>>(qkv_w, WbT, 512, 1536);
  k_transpose_bf16<<<1024, 256, 0, stream>>>(proj_w, PbT, 512, 512);
  k_qkv_gemm<<<dim3(12, 722), 256, 0, stream>>>(Xb, WbT, Q, K, Vt);
  k_attn<<<dim3(6, 8, 256), 256, ATTN_SMEM, stream>>>(Q, K, Vt, bias_table, O);
  k_proj_gemm<<<dim3(4, 722), 256, 0, stream>>>(O, PbT, proj_b, out);
}
